// _ShiftedWindowAttention_82549271429455
// MI455X (gfx1250) — matrix operand order repaired and checked by reading, not compiled, not run on hardware
//
#include <hip/hip_runtime.h>
#include <hip/hip_bf16.h>

// ---- fused Swin shifted-window attention for gfx1250 (MI455X) ----
// One workgroup (256 threads = 8 wave32) per 7x7 window. All intermediates in
// dynamic LDS (~303 KB of the 320 KB WGP LDS). bf16 WMMA (16x16x32, f32 acc).
// Phase-1 gather uses GLOBAL_LOAD_ASYNC_TO_LDS_B32 (ASYNCcnt). Weights are
// pre-packed to bf16 in d_ws by a prep kernel (one pass vs 4096 re-converts).

typedef __attribute__((ext_vector_type(16))) __bf16 v16bf;
typedef __attribute__((ext_vector_type(2)))  __bf16 v2bf;
typedef __attribute__((ext_vector_type(8)))  float  v8f;

#define C_    192
#define H_    448
#define W_    448
#define NH_   6
#define HD_   32
#define WS_   7
#define SH_   3
#define N_    49
#define NWX   64
#define MPAD  64

#define AS_A  196   // a_in row stride (elements), even
#define QS_   584   // qkv row stride (ushort), even
#define SS_   66    // score/prob/vT row stride, even
#define PS_   200   // attn-out row stride (ushort), even

#if __has_builtin(__builtin_amdgcn_global_load_async_to_lds_b32)
#define HAS_ASYNC_LDS 1
typedef __attribute__((address_space(1))) int gint_t;   // global-side param type
typedef __attribute__((address_space(3))) int lint_t;   // LDS-side param type
#else
#define HAS_ASYNC_LDS 0
#endif

union AF { v16bf v; unsigned int u[8]; };
union CF { v8f  v; float f[8]; };

__device__ __forceinline__ unsigned int pack2(float lo, float hi) {
#if __has_builtin(__builtin_amdgcn_cvt_pk_bf16_f32)
  union { v2bf v; unsigned int u; } t;
  t.v = __builtin_amdgcn_cvt_pk_bf16_f32(lo, hi);
  return t.u;
#else
  // round-half-up bf16 pack: 1 add per float + 1 merge (clang emits v_perm)
  unsigned a = __float_as_uint(lo) + 0x8000u;
  unsigned b = __float_as_uint(hi) + 0x8000u;
  return (a >> 16) | (b & 0xFFFF0000u);
#endif
}
__device__ __forceinline__ unsigned short f2bf(float f) {
  return (unsigned short)((__float_as_uint(f) + 0x8000u) >> 16);
}
__device__ __forceinline__ void wait_async0() {
#if __has_builtin(__builtin_amdgcn_s_wait_asynccnt)
  __builtin_amdgcn_s_wait_asynccnt(0);
#else
  asm volatile("s_wait_asynccnt 0x0" ::: "memory");
#endif
}
// A-matrix (16x32 bf16) per-lane K offset for VGPR v, lane-half h (contiguous: k = 16h + 2v, matches B)
__device__ __forceinline__ int akoff(int v, int h) {
  return h * 16 + 2 * v;
}

// ---- weight pre-pack: f32 -> bf16 pairs in d_ws ----
__global__ __launch_bounds__(256) void weight_prep_kernel(
    const float* __restrict__ qkvw, const float* __restrict__ projw,
    unsigned int* __restrict__ wq, unsigned int* __restrict__ wp)
{
  const int NQ = (3 * C_ * C_) / 2;   // 55296 pairs
  const int NP = (C_ * C_) / 2;       // 18432 pairs
  int i = blockIdx.x * 256 + threadIdx.x;
  if (i < NQ) {
    float2 p = ((const float2*)qkvw)[i];
    wq[i] = pack2(p.x, p.y);
  } else if (i < NQ + NP) {
    int j = i - NQ;
    float2 p = ((const float2*)projw)[j];
    wp[j] = pack2(p.x, p.y);
  }
}

template <bool PREW>
__global__ __launch_bounds__(256) void swin_fused_kernel(
    const float* __restrict__ x,
    const float* __restrict__ qkvw,
    const float* __restrict__ qkvb,
    const float* __restrict__ projw,
    const float* __restrict__ projb,
    const float* __restrict__ btab,
    const unsigned short* __restrict__ wqbf,
    const unsigned short* __restrict__ wpbf,
    float* __restrict__ out)
{
  extern __shared__ unsigned char smem[];
  unsigned short* a_in  = (unsigned short*)smem;                    // 64*196 bf16
  unsigned short* qkv   = a_in + MPAD * AS_A;                       // 64*584 bf16
  float*          sc    = (float*)(qkv + MPAD * QS_);               // 6*64*66 f32
  unsigned short* probs = (unsigned short*)(sc + NH_ * MPAD * SS_); // 6*64*66 bf16
  unsigned short* po    = probs + NH_ * MPAD * SS_;                 // 64*200 bf16
  unsigned short* vT    = po + MPAD * PS_;                          // 6*32*66 bf16

  const int tid  = threadIdx.x;
  const int lane = tid & 31;
  const int wid  = tid >> 5;
  const int half = lane >> 4;
  const int ln16 = lane & 15;

  const int wi = blockIdx.x;
  const int wr = wi / NWX, wc = wi % NWX;

  // ---- Phase 1: gather shifted window tokens -> bf16 LDS (rows 49..63 = 0) ----
#if HAS_ASYNC_LDS
  {
    float* stage = sc;  // reuse score region as f32 staging (dead until phase 3)
    for (int i = tid; i < N_ * C_; i += 256) {
      int m = i / C_, c = i % C_;
      int tr = m / WS_, tc = m % WS_;
      int gh = wr * WS_ + tr + SH_; if (gh >= H_) gh -= H_;
      int gw = wc * WS_ + tc + SH_; if (gw >= W_) gw -= W_;
      __builtin_amdgcn_global_load_async_to_lds_b32(
          (gint_t*)(x + (size_t)c * (H_ * W_) + gh * W_ + gw),
          (lint_t*)&stage[m * AS_A + c], 0, 0);
    }
    wait_async0();
    __syncthreads();
    for (int i = tid; i < MPAD * (C_ / 2); i += 256) {
      int m = i / (C_ / 2), c = 2 * (i % (C_ / 2));
      float lo = 0.f, hi = 0.f;
      if (m < N_) { float2 p = *(const float2*)&stage[m * AS_A + c]; lo = p.x; hi = p.y; }
      *(unsigned int*)&a_in[m * AS_A + c] = pack2(lo, hi);
    }
  }
#else
  for (int i = tid; i < MPAD * C_; i += 256) {
    int m = i / C_, c = i % C_;
    float val = 0.f;
    if (m < N_) {
      int tr = m / WS_, tc = m % WS_;
      int gh = wr * WS_ + tr + SH_; if (gh >= H_) gh -= H_;
      int gw = wc * WS_ + tc + SH_; if (gw >= W_) gw -= W_;
      val = x[(size_t)c * (H_ * W_) + gh * W_ + gw];
    }
    a_in[m * AS_A + c] = f2bf(val);
  }
#endif
  for (int i = tid; i < NH_ * MPAD * SS_; i += 256) probs[i] = 0;
  __syncthreads();

  // ---- Phase 2: qkv = a_in @ qkvw^T + qkvb (M=64, K=192, Nc=576) ----
  const float qscale = 0.17677669529663687f; // 1/sqrt(32)
  for (int job = wid; job < 4 * 36; job += 8) {
    int mt = job & 3, nt = job >> 2;
    CF acc;
    #pragma unroll
    for (int i = 0; i < 8; ++i) acc.f[i] = 0.f;
    int am = mt * 16 + ln16;
    int ng = nt * 16 + ln16;
    #pragma unroll
    for (int kt = 0; kt < 6; ++kt) {
      AF a, b;
      #pragma unroll
      for (int v = 0; v < 8; ++v) {
        int ka = kt * 32 + akoff(v, half);
        a.u[v] = *(const unsigned int*)&a_in[am * AS_A + ka];
        int kb = kt * 32 + half * 16 + 2 * v;
        if constexpr (PREW) {
          b.u[v] = *(const unsigned int*)&wqbf[(size_t)ng * C_ + kb];
        } else {
          float2 wv = *(const float2*)&qkvw[(size_t)ng * C_ + kb];
          b.u[v] = pack2(wv.x, wv.y);
        }
      }
      acc.v = __builtin_amdgcn_wmma_f32_16x16x32_bf16(
          false, a.v, false, b.v, (short)0, acc.v, false, false);
    }
    float bias = qkvb[ng];
    float scl = (ng < C_) ? qscale : 1.f;
    #pragma unroll
    for (int r = 0; r < 8; ++r) {
      int m = mt * 16 + half * 8 + r;
      unsigned short bfv = f2bf((acc.f[r] + bias) * scl);
      qkv[m * QS_ + ng] = bfv;
      if (ng >= 2 * C_) {                         // v region: transposed copy
        int d = ng - 2 * C_;                      // d = nh*32 + dd
        vT[d * SS_ + m] = bfv;
      }
    }
  }
  __syncthreads();

  // ---- Phase 3: scores = q @ k^T + rel_bias + shift_mask, per head ----
  for (int job = wid; job < NH_ * 16; job += 8) {
    int nh = job / 16, mt = (job / 4) & 3, nt = job & 3;
    AF a, b;
    int am = mt * 16 + ln16;
    int bn = nt * 16 + ln16;
    #pragma unroll
    for (int v = 0; v < 8; ++v) {
      int ka = akoff(v, half);                 // d in 0..31
      a.u[v] = *(const unsigned int*)&qkv[am * QS_ + nh * HD_ + ka];
      int kb = half * 16 + 2 * v;              // d
      b.u[v] = *(const unsigned int*)&qkv[bn * QS_ + C_ + nh * HD_ + kb];
    }
    CF acc;
    #pragma unroll
    for (int i = 0; i < 8; ++i) acc.f[i] = 0.f;
    acc.v = __builtin_amdgcn_wmma_f32_16x16x32_bf16(
        false, a.v, false, b.v, (short)0, acc.v, false, false);
    int j = nt * 16 + ln16;
    #pragma unroll
    for (int r = 0; r < 8; ++r) {
      int i = mt * 16 + half * 8 + r;
      float s = acc.f[r];
      if (i < N_ && j < N_) {
        int ri = i / WS_, ci = i % WS_, rj = j / WS_, cj = j % WS_;
        int rel = (ri - rj) + (ci - cj);
        if (rel < 0) rel += 169;               // reference's wraparound gather
        s += btab[rel * NH_ + nh];
        int gri = wr * WS_ + ri, gci = wc * WS_ + ci;
        int grj = wr * WS_ + rj, gcj = wc * WS_ + cj;
        int mi = ((gri >= H_ - WS_) + (gri >= H_ - SH_)) * 3 +
                  (gci >= W_ - WS_) + (gci >= W_ - SH_);
        int mj = ((grj >= H_ - WS_) + (grj >= H_ - SH_)) * 3 +
                  (gcj >= W_ - WS_) + (gcj >= W_ - SH_);
        if (mi != mj) s -= 100.f;
      }
      sc[nh * (MPAD * SS_) + i * SS_ + j] = s;
    }
  }
  __syncthreads();

  // ---- Phase 4: softmax over j<49 (3-pass, per-thread rows) -> bf16 probs ----
  for (int r = tid; r < NH_ * N_; r += 256) {
    int nh = r / N_, i = r % N_;
    const float* srow = &sc[nh * (MPAD * SS_) + i * SS_];
    float mx = -1e30f;
    for (int j = 0; j < N_; ++j) mx = fmaxf(mx, srow[j]);
    float sum = 0.f;
    for (int j = 0; j < N_; ++j) sum += __expf(srow[j] - mx);
    float inv = 1.f / sum;
    unsigned short* prow = &probs[nh * (MPAD * SS_) + i * SS_];
    for (int j = 0; j < N_; ++j) prow[j] = f2bf(__expf(srow[j] - mx) * inv);
  }
  __syncthreads();

  // ---- Phase 5: attn_out = probs @ v (per head, M=64,K=64,N=32) ----
  for (int job = wid; job < NH_ * 8; job += 8) {
    int nh = job / 8, mt = (job / 2) & 3, nt = job & 1;
    CF acc;
    #pragma unroll
    for (int i = 0; i < 8; ++i) acc.f[i] = 0.f;
    int am = mt * 16 + ln16;
    int d  = nt * 16 + ln16;                   // 0..31 within head
    #pragma unroll
    for (int kt = 0; kt < 2; ++kt) {
      AF a, b;
      #pragma unroll
      for (int v = 0; v < 8; ++v) {
        int ka = kt * 32 + akoff(v, half);
        a.u[v] = *(const unsigned int*)&probs[nh * (MPAD * SS_) + am * SS_ + ka];
        int kb = kt * 32 + half * 16 + 2 * v;  // token pair (j, j+1)
        b.u[v] = *(const unsigned int*)&vT[(nh * HD_ + d) * SS_ + kb];
      }
      acc.v = __builtin_amdgcn_wmma_f32_16x16x32_bf16(
          false, a.v, false, b.v, (short)0, acc.v, false, false);
    }
    #pragma unroll
    for (int r = 0; r < 8; ++r) {
      int m = mt * 16 + half * 8 + r;
      po[m * PS_ + nh * HD_ + d] = f2bf(acc.f[r]);
    }
  }
  __syncthreads();

  // ---- Phase 6: out = po @ projw^T + projb, scatter with reverse shift ----
  for (int job = wid; job < 4 * 12; job += 8) {
    int mt = job & 3, nt = job >> 2;
    CF acc;
    #pragma unroll
    for (int i = 0; i < 8; ++i) acc.f[i] = 0.f;
    int am = mt * 16 + ln16;
    int ng = nt * 16 + ln16;
    #pragma unroll
    for (int kt = 0; kt < 6; ++kt) {
      AF a, b;
      #pragma unroll
      for (int v = 0; v < 8; ++v) {
        int ka = kt * 32 + akoff(v, half);
        a.u[v] = *(const unsigned int*)&po[am * PS_ + ka];
        int kb = kt * 32 + half * 16 + 2 * v;
        if constexpr (PREW) {
          b.u[v] = *(const unsigned int*)&wpbf[(size_t)ng * C_ + kb];
        } else {
          float2 wv = *(const float2*)&projw[(size_t)ng * C_ + kb];
          b.u[v] = pack2(wv.x, wv.y);
        }
      }
      acc.v = __builtin_amdgcn_wmma_f32_16x16x32_bf16(
          false, a.v, false, b.v, (short)0, acc.v, false, false);
    }
    float pb = projb[ng];
    #pragma unroll
    for (int r = 0; r < 8; ++r) {
      int t = mt * 16 + half * 8 + r;
      if (t < N_) {
        int tr = t / WS_, tc = t % WS_;
        // reverse window partition + reverse roll == forward gather address
        int gh = wr * WS_ + tr + SH_; if (gh >= H_) gh -= H_;
        int gw = wc * WS_ + tc + SH_; if (gw >= W_) gw -= W_;
        out[(size_t)ng * (H_ * W_) + gh * W_ + gw] = acc.f[r] + pb;
      }
    }
  }
}

extern "C" void kernel_launch(void* const* d_in, const int* in_sizes, int n_in,
                              void* d_out, int out_size, void* d_ws, size_t ws_size,
                              hipStream_t stream) {
  const float* x     = (const float*)d_in[0];
  const float* qkvw  = (const float*)d_in[1];
  const float* qkvb  = (const float*)d_in[2];
  const float* projw = (const float*)d_in[3];
  const float* projb = (const float*)d_in[4];
  const float* btab  = (const float*)d_in[5];
  float* out = (float*)d_out;

  const size_t smem = (size_t)MPAD * AS_A * 2        // a_in (bf16)
                    + (size_t)MPAD * QS_ * 2         // qkv (bf16)
                    + (size_t)NH_ * MPAD * SS_ * 4   // scores (f32) / f32 staging
                    + (size_t)NH_ * MPAD * SS_ * 2   // probs (bf16)
                    + (size_t)MPAD * PS_ * 2         // attn-out (bf16)
                    + (size_t)NH_ * HD_ * SS_ * 2;   // vT (bf16)

  const size_t wq_bytes = (size_t)3 * C_ * C_ * 2;   // 221184
  const size_t wp_bytes = (size_t)C_ * C_ * 2;       // 73728
  const bool prew = ws_size >= wq_bytes + wp_bytes;

  dim3 grid(NWX * NWX), block(256);
  if (prew) {
    unsigned int* wq = (unsigned int*)d_ws;
    unsigned int* wp = (unsigned int*)((char*)d_ws + wq_bytes);
    int pairs = (3 * C_ * C_ + C_ * C_) / 2;         // 73728
    weight_prep_kernel<<<(pairs + 255) / 256, 256, 0, stream>>>(qkvw, projw, wq, wp);
    (void)hipFuncSetAttribute(reinterpret_cast<const void*>(&swin_fused_kernel<true>),
                              hipFuncAttributeMaxDynamicSharedMemorySize, (int)smem);
    swin_fused_kernel<true><<<grid, block, smem, stream>>>(
        x, qkvw, qkvb, projw, projb, btab,
        (const unsigned short*)wq, (const unsigned short*)wp, out);
  } else {
    (void)hipFuncSetAttribute(reinterpret_cast<const void*>(&swin_fused_kernel<false>),
                              hipFuncAttributeMaxDynamicSharedMemorySize, (int)smem);
    swin_fused_kernel<false><<<grid, block, smem, stream>>>(
        x, qkvw, qkvb, projw, projb, btab, nullptr, nullptr, out);
  }
}
